// KnnLinearFixModel_17368847745070
// MI455X (gfx1250) — matrix operand order repaired and checked by reading, not compiled, not run on hardware
//
#include <hip/hip_runtime.h>
#include <hip/hip_bf16.h>

// ---------------------------------------------------------------------------
// Collapsed TTT (train-at-test-time linear head) for MI455X / gfx1250.
//
//   W_{t+1} = a*W_t - c*diff_t^T X     (a = 1-LR*WD, c = LR*2/(K*L))
//   O_t = X W_t^T  =>  O_{t+1} = a*O_t - c*G*diff_t,  G = X X^T (constant!)
//   b_{t+1} = a*b_t - c*colsum(diff_t)
//   logits  = a^20*(W0 e) - c*sum_t a^{19-t} diff_t^T (X e) + b_20
//
// One workgroup (4 wave32) per batch element; X(f16), G(f16), diff^T(f16) in
// LDS; O held in v_wmma f32 accumulators (one 16x16 tile per wave).
// ---------------------------------------------------------------------------

typedef __attribute__((ext_vector_type(16))) _Float16 v16h;
typedef __attribute__((ext_vector_type(8)))  _Float16 v8h;
typedef __attribute__((ext_vector_type(4)))  _Float16 v4h;
typedef __attribute__((ext_vector_type(8)))  float    v8f;

#define KDEM 64      // K demos
#define DDIM 1024    // embedding dim
#define LPAD 16      // L padded 8 -> 16 for WMMA N
#define XS   1032    // Xh / W0h row stride in halfs (pad: 2064B rows, conflict-free)
#define GS   72      // Gh / Dh row stride in halfs (144B rows, conflict-free)

static __device__ inline v8f wmma16(v16h a, v16h b, v8f c) {
  // D = A(16x32,f16) * B(32x16,f16) + C(16x16,f32)
  return __builtin_amdgcn_wmma_f32_16x16x32_f16(false, a, false, b, (short)0, c,
                                                false, false);
}

// A fragment (16x32, row m = row_base + lane%16) from row-major [m][k] array.
// ISA A layout: lane-half kh holds K in {kh*8..kh*8+7} U {16+kh*8..16+kh*8+7}.
static __device__ inline v16h frag_a(const _Float16* base, int row_base,
                                     int stride, int k0, int lane) {
  const _Float16* p = base + (row_base + (lane & 15)) * stride;
  int kh = lane >> 4;
  v8h lo = *(const v8h*)(p + k0 + kh * 8);        // K = k0 + kh*8 + 0..7
  v8h hi = *(const v8h*)(p + k0 + 16 + kh * 8);   // K = k0 + 16 + kh*8 + 0..7
  return __builtin_shufflevector(lo, hi, 0, 1, 2, 3, 4, 5, 6, 7,
                                 8, 9, 10, 11, 12, 13, 14, 15);
}

// B fragment (32x16, col n = row_base + lane%16) from the TRANSPOSED row-major
// array [n][k].  B layout: lane-half kh holds K in {kh*8..kh*8+7} U {16+kh*8..16+kh*8+7}.
static __device__ inline v16h frag_b(const _Float16* baseT, int row_base,
                                     int stride, int k0, int lane) {
  const _Float16* p = baseT + (row_base + (lane & 15)) * stride + k0 + (lane >> 4) * 8;
  v8h lo = *(const v8h*)(p);
  v8h hi = *(const v8h*)(p + 16);
  return __builtin_shufflevector(lo, hi, 0, 1, 2, 3, 4, 5, 6, 7,
                                 8, 9, 10, 11, 12, 13, 14, 15);
}

__global__ __launch_bounds__(128)
void ttt_collapsed_kernel(const float* __restrict__ demo,   // [B,64,1024]
                          const float* __restrict__ emb,    // [B,1024]
                          const float* __restrict__ W0,     // [B,8,1024]
                          const float* __restrict__ b0,     // [B,8]
                          const int*   __restrict__ labels, // [B,64]
                          float* __restrict__ out)          // [B,8]
{
  extern __shared__ __align__(16) char smem[];
  _Float16* Xh  = (_Float16*)smem;       // [64][XS]  f16 X          132096 B
  _Float16* W0h = Xh + KDEM * XS;        // [16][XS]  f16 W0 (pad 0)  33024 B
  _Float16* Gh  = W0h + LPAD * XS;       // [64][GS]  f16 G = X X^T    9216 B
  _Float16* Dh  = Gh + KDEM * GS;        // [16][GS]  f16 diff^T       2304 B
  float* se  = (float*)(Dh + LPAD * GS); // [1024] query e             4096 B
  float* sv  = se + DDIM;                // [64]   s = X e              256 B
  float* ba  = sv + KDEM;                // [16]   bias b_t              64 B
  float* uv  = ba + LPAD;                // [16]   logit accumulator     64 B
  float* we  = uv + LPAD;                // [16]   W0 e                  64 B
  int*   lab = (int*)(we + LPAD);        // [64]   labels               256 B

  const int b    = blockIdx.x;
  const int tid  = threadIdx.x;
  const int w    = tid >> 5;    // wave id 0..3
  const int lane = tid & 31;

  // ---- phase 0: stage inputs into LDS (f32 -> f16 for matmul operands) ----
  const float* Xg = demo + (size_t)b * KDEM * DDIM;
  for (int row = w; row < KDEM; row += 4) {
    const float4* src = (const float4*)(Xg + row * DDIM);
    _Float16* dst = Xh + row * XS;
    for (int i = lane; i < DDIM / 4; i += 32) {
      float4 v = src[i];
      v4h h = {(_Float16)v.x, (_Float16)v.y, (_Float16)v.z, (_Float16)v.w};
      *(v4h*)(dst + i * 4) = h;
    }
  }
  const float* Wg = W0 + (size_t)b * 8 * DDIM;
  for (int row = w; row < LPAD; row += 4) {
    _Float16* dst = W0h + row * XS;
    if (row < 8) {
      const float4* src = (const float4*)(Wg + row * DDIM);
      for (int i = lane; i < DDIM / 4; i += 32) {
        float4 v = src[i];
        v4h h = {(_Float16)v.x, (_Float16)v.y, (_Float16)v.z, (_Float16)v.w};
        *(v4h*)(dst + i * 4) = h;
      }
    } else {
      v4h z = {(_Float16)0.f, (_Float16)0.f, (_Float16)0.f, (_Float16)0.f};
      for (int i = lane; i < DDIM / 4; i += 32) *(v4h*)(dst + i * 4) = z;
    }
  }
  const float* eg = emb + (size_t)b * DDIM;
  for (int i = tid; i < DDIM; i += 128) se[i] = eg[i];
  if (tid < KDEM) lab[tid] = labels[(size_t)b * KDEM + tid];
  if (tid < LPAD) {
    ba[tid] = (tid < 8) ? b0[(size_t)b * 8 + tid] : 0.f;
    uv[tid] = 0.f;
  }
  __syncthreads();

  // ---- phase 1: G = X X^T (64x64, contraction over D=1024), 16 tiles ----
  for (int t = w; t < 16; t += 4) {
    const int mi = t >> 2, ni = t & 3;
    v8f acc = {};
#pragma unroll 4
    for (int kf = 0; kf < DDIM / 32; ++kf) {
      v16h a  = frag_a(Xh, mi * 16, XS, kf * 32, lane);
      v16h bb = frag_b(Xh, ni * 16, XS, kf * 32, lane);  // B = X^T
      acc = wmma16(a, bb, acc);
    }
    const int n = ni * 16 + (lane & 15);
    const int mb = mi * 16 + (lane >> 4) * 8;
#pragma unroll
    for (int r = 0; r < 8; ++r)
      Gh[(mb + r) * GS + n] = (_Float16)acc[r];   // C/D layout: M=r+8*hi, N=lane%16
  }

  // ---- phase 2: O0 = X W0^T (each wave owns one 16-row M tile of O) ----
  v8f accO = {};
#pragma unroll 4
  for (int kf = 0; kf < DDIM / 32; ++kf) {
    v16h a  = frag_a(Xh, w * 16, XS, kf * 32, lane);
    v16h bb = frag_b(W0h, 0, XS, kf * 32, lane);         // B = W0^T
    accO = wmma16(a, bb, accO);
  }

  // ---- phase 3: s = X e, we = W0 e (f32 accumulation) ----
  if (tid < KDEM) {
    float acc = 0.f;
    const _Float16* xr = Xh + tid * XS;
    for (int d = 0; d < DDIM; ++d) acc += (float)xr[d] * se[d];
    sv[tid] = acc;
  } else if (tid < KDEM + LPAD) {
    const int l = tid - KDEM;
    float acc = 0.f;
    const _Float16* wr = W0h + l * XS;
    for (int d = 0; d < DDIM; ++d) acc += (float)wr[d] * se[d];
    we[l] = acc;
  }
  __syncthreads();

  const float alpha = 1.0f - 1e-4f * 0.01f;                 // 1 - LR*WD
  const float cfac  = 1e-4f * (2.0f / (64.0f * 8.0f));      // LR * 2/(K*L)

  // ---- 20 SGD steps, entirely in the 64x16 O-space ----
  for (int step = 0; step < 20; ++step) {
    // (a) diff_t = O + 1 b^T - Y  -> Dh (stored transposed: [l][demo])
    {
      const int l  = lane & 15;
      const int mb = w * 16 + (lane >> 4) * 8;
      const float bl = ba[l];
#pragma unroll
      for (int r = 0; r < 8; ++r) {
        const int m = mb + r;
        const float y = (lab[m] == l) ? 1.f : 0.f;
        Dh[l * GS + m] = (_Float16)(accO[r] + bl - y);
      }
    }
    __syncthreads();

    // (b) bias + logit-accumulator update (16 lanes of wave 0)
    if (w == 0 && lane < LPAD) {
      float cs = 0.f, dt = 0.f;
      const _Float16* dr = Dh + lane * GS;
      for (int k = 0; k < KDEM; ++k) {
        const float dv = (float)dr[k];
        cs += dv;
        dt += dv * sv[k];
      }
      ba[lane] = alpha * ba[lane] - cfac * cs;    // b_{t+1}
      uv[lane] = alpha * uv[lane] + dt;           // sum_t a^{T-1-t} diff^T s
    }

    // (c) P = G @ diff (64x16, contraction over 64 demos), O = a*O - c*P
    v8f pac = {};
#pragma unroll
    for (int kf = 0; kf < KDEM / 32; ++kf) {
      v16h a  = frag_a(Gh, w * 16, GS, kf * 32, lane);
      v16h bb = frag_b(Dh, 0, GS, kf * 32, lane);        // B = diff (Dh = diff^T)
      pac = wmma16(a, bb, pac);
    }
#pragma unroll
    for (int r = 0; r < 8; ++r) accO[r] = alpha * accO[r] - cfac * pac[r];
    __syncthreads();
  }

  // ---- epilogue: logits = a^20 * (W0 e) - c * u + b_20 (first 8 cols) ----
  if (tid < 8) {
    float aT = 1.f;
#pragma unroll
    for (int t = 0; t < 20; ++t) aT *= alpha;
    out[(size_t)b * 8 + tid] = aT * we[tid] - cfac * uv[tid] + ba[tid];
  }
}

extern "C" void kernel_launch(void* const* d_in, const int* in_sizes, int n_in,
                              void* d_out, int out_size, void* d_ws, size_t ws_size,
                              hipStream_t stream) {
  (void)n_in; (void)d_ws; (void)ws_size; (void)out_size;
  const float* demo   = (const float*)d_in[0];
  const float* emb    = (const float*)d_in[1];
  const float* W0     = (const float*)d_in[2];
  const float* b0     = (const float*)d_in[3];
  const int*   labels = (const int*)d_in[4];
  float* out = (float*)d_out;

  const int B = in_sizes[1] / DDIM;   // 512

  const size_t shmem =
      (size_t)(KDEM * XS + LPAD * XS + KDEM * GS + LPAD * GS) * sizeof(_Float16)
      + (size_t)(DDIM + KDEM + 3 * LPAD) * sizeof(float)
      + (size_t)KDEM * sizeof(int);

  (void)hipFuncSetAttribute((const void*)ttt_collapsed_kernel,
                            hipFuncAttributeMaxDynamicSharedMemorySize,
                            (int)shmem);
  ttt_collapsed_kernel<<<B, 128, shmem, stream>>>(demo, emb, W0, b0, labels, out);
}
